// DyDCNv2_695784702274
// MI455X (gfx1250) — matrix operand order repaired and checked by reading, not compiled, not run on hardware
//
#include <hip/hip_runtime.h>
#include <hip/hip_bf16.h>
#include <math.h>

// ---------------- problem constants ----------------
#define BB   8
#define CC   64
#define OO   64
#define KW   3
#define KK   9          // KW*KW
#define HH   128
#define WW   128
#define HWP  (HH*WW)            // 16384 pixels per image
#define NPIX (BB*HWP)           // 131072 total pixels
#define CK   (KK*CC)            // 576 = GEMM K dim  (ordered kk*64 + c)
#define KCH  18                 // 576 / 32 WMMA K-chunks
#define EPSV 1e-5f

typedef __attribute__((ext_vector_type(16))) _Float16 v16h;
typedef __attribute__((ext_vector_type(8)))  _Float16 v8h;
typedef __attribute__((ext_vector_type(8)))  float    v8f;

// ---------------- workspace layout (bytes) ----------------
#define WS_XH    0u                       // NHWC f16 x:          16,777,216
#define WS_OM    16777216u                // per-pixel 27 f32:    14,155,776
#define WS_BF    30932992u                // B fragments f16:         73,728
#define WS_Y     31006720u                // y (pixel-major) f32: 33,554,432
#define WS_STAT  64561152u                // 128 floats sum/sumsq

// =============================================================
// 0) zero the BN statistics accumulators
// =============================================================
__global__ void k_init_stats(float* stats) {
    if (threadIdx.x < 128) stats[threadIdx.x] = 0.0f;
}

// =============================================================
// 1) NCHW f32 -> NHWC f16  (write-coalesced)
// =============================================================
__global__ void k_nhwc(const float* __restrict__ x, _Float16* __restrict__ xh) {
    int idx = blockIdx.x * blockDim.x + threadIdx.x;     // over B*HW*C
    int c    = idx & 63;
    int rest = idx >> 6;
    int hw   = rest & (HWP - 1);
    int b    = rest >> 14;
    xh[idx] = (_Float16)x[(((size_t)b * CC + c) << 14) + hw];
}

// =============================================================
// 2) offset/mask conv: 27ch 3x3 conv, w_om cached in LDS.
//    Stores per-pixel [dy0,dx0,dy1,dx1,...,dy8,dx8, m0..m8(sigmoid)]
// =============================================================
__global__ void k_offconv(const float* __restrict__ x,
                          const float* __restrict__ w_om,
                          const float* __restrict__ b_om,
                          float* __restrict__ om) {
    __shared__ float wl[27 * CC * KK];    // 15552 floats = 62208 B
    __shared__ float bl[27];
    for (int i = threadIdx.x; i < 27 * CC * KK; i += blockDim.x) wl[i] = w_om[i];
    if (threadIdx.x < 27) bl[threadIdx.x] = b_om[threadIdx.x];
    __syncthreads();

    int p  = blockIdx.x * blockDim.x + threadIdx.x;      // pixel id
    int b  = p >> 14;
    int hw = p & (HWP - 1);
    int h  = hw >> 7;
    int w  = hw & (WW - 1);

    float acc[27];
#pragma unroll
    for (int o = 0; o < 27; ++o) acc[o] = bl[o];

    const float* xb = x + ((size_t)b * CC << 14);
    for (int c = 0; c < CC; ++c) {
        const float* xc = xb + ((size_t)c << 14);
        for (int di = 0; di < 3; ++di) {
            int hh2 = h + di - 1;
            bool rv = (hh2 >= 0) && (hh2 < HH);
            for (int dj = 0; dj < 3; ++dj) {
                int ww2 = w + dj - 1;
                float xv = 0.0f;
                if (rv && ww2 >= 0 && ww2 < WW) xv = xc[hh2 * WW + ww2];
                int wbase = c * KK + di * 3 + dj;
#pragma unroll
                for (int o = 0; o < 27; ++o)
                    acc[o] = fmaf(wl[o * (CC * KK) + wbase], xv, acc[o]);
            }
        }
    }

    float* op = om + (size_t)p * 27;
#pragma unroll
    for (int i = 0; i < 18; ++i) op[i] = acc[i];
#pragma unroll
    for (int i = 18; i < 27; ++i) op[i] = 1.0f / (1.0f + expf(-acc[i]));
}

// =============================================================
// 3) pack weight (O,C,3,3) into per-lane WMMA B fragments, f16.
//    frag idx = ((n_tile*18 + k_chunk)*32 + lane)*16 + h
//    lane<16: n=lane,    k = kc*32 + h
//    lane>=16: n=lane-16, k = kc*32 + 16 + h
//    K-dim order: k = kk*64 + c
// =============================================================
__global__ void k_packB(const float* __restrict__ weight, _Float16* __restrict__ bf) {
    int idx = blockIdx.x * blockDim.x + threadIdx.x;     // 4*18*32*16 = 36864
    if (idx >= 4 * KCH * 32 * 16) return;
    int nt   = idx / (KCH * 32 * 16);
    int r    = idx % (KCH * 32 * 16);
    int kc   = r / 512;
    int r2   = r % 512;
    int lane = r2 >> 4;
    int hh2  = r2 & 15;
    int n = nt * 16 + (lane & 15);
    int k = kc * 32 + hh2 + ((lane >= 16) ? 16 : 0);
    int c  = k & 63;
    int kk = k >> 6;
    bf[idx] = (_Float16)weight[((size_t)n * CC + c) * KK + kk];
}

// =============================================================
// 4) main deformable-GEMM kernel.
//    Workgroup: 256 threads (8 waves), 32 pixels x 64 outputs.
//    Phase 1: build A tile (32 x 576 f16) in LDS via bilinear
//             sampling of NHWC x (one 64-ch vector per (pix,tap)).
//    Phase 2: each wave -> one 16x16 tile, 18 x v_wmma_f32_16x16x32_f16.
// =============================================================
__global__ void k_main(const _Float16* __restrict__ xh,
                       const float* __restrict__ om,
                       const _Float16* __restrict__ bfrag,
                       const float* __restrict__ bias,
                       float* __restrict__ y) {
    constexpr int ASTRIDE = CK + 8;                      // 584 halves, 16B-aligned rows
    __shared__ _Float16 sA[32][ASTRIDE];                 // 37,376 B

    const int tid = threadIdx.x;
    const int p0  = blockIdx.x * 32;

    // ---------- phase 1: sample A ----------
    for (int task = tid; task < 32 * KK; task += blockDim.x) {
        int pl = task / KK;
        int kk = task % KK;
        int p  = p0 + pl;
        int b  = p >> 14;
        int hw = p & (HWP - 1);
        int h  = hw >> 7;
        int w  = hw & (WW - 1);

        const float* op = om + (size_t)p * 27;
        float dy = op[2 * kk];
        float dx = op[2 * kk + 1];
        float mk = op[18 + kk];

        float py = (float)h - 1.0f + (float)(kk / 3) + dy;
        float px = (float)w - 1.0f + (float)(kk % 3) + dx;
        float fy = floorf(py), fx = floorf(px);
        float ly = py - fy,    lx = px - fx;
        int iy0 = (int)fy, ix0 = (int)fx;
        int iy1 = iy0 + 1, ix1 = ix0 + 1;

        float v00 = (iy0 >= 0 && iy0 < HH && ix0 >= 0 && ix0 < WW) ? 1.0f : 0.0f;
        float v01 = (iy0 >= 0 && iy0 < HH && ix1 >= 0 && ix1 < WW) ? 1.0f : 0.0f;
        float v10 = (iy1 >= 0 && iy1 < HH && ix0 >= 0 && ix0 < WW) ? 1.0f : 0.0f;
        float v11 = (iy1 >= 0 && iy1 < HH && ix1 >= 0 && ix1 < WW) ? 1.0f : 0.0f;
        float w00 = (1.0f - ly) * (1.0f - lx) * mk * v00;
        float w01 = (1.0f - ly) * lx          * mk * v01;
        float w10 = ly          * (1.0f - lx) * mk * v10;
        float w11 = ly          * lx          * mk * v11;

        int cy0 = min(max(iy0, 0), HH - 1), cx0 = min(max(ix0, 0), WW - 1);
        int cy1 = min(max(iy1, 0), HH - 1), cx1 = min(max(ix1, 0), WW - 1);
        size_t ib = (size_t)b << 14;
        const v8h* r00 = (const v8h*)(xh + ((ib + cy0 * WW + cx0) << 6));
        const v8h* r01 = (const v8h*)(xh + ((ib + cy0 * WW + cx1) << 6));
        const v8h* r10 = (const v8h*)(xh + ((ib + cy1 * WW + cx0) << 6));
        const v8h* r11 = (const v8h*)(xh + ((ib + cy1 * WW + cx1) << 6));

        _Float16* dst = &sA[pl][kk * CC];
#pragma unroll
        for (int cc = 0; cc < 8; ++cc) {                 // 8 halves per chunk
            v8h a = r00[cc], bq = r01[cc], cq = r10[cc], dq = r11[cc];
            v8h o;
#pragma unroll
            for (int i = 0; i < 8; ++i)
                o[i] = (_Float16)((float)a[i]  * w00 + (float)bq[i] * w01 +
                                  (float)cq[i] * w10 + (float)dq[i] * w11);
            *(v8h*)(dst + cc * 8) = o;
        }
    }
    __syncthreads();

    // ---------- phase 2: WMMA ----------
    const int wave  = tid >> 5;
    const int lane  = tid & 31;
    const int m_sub = (wave >> 2) * 16;                  // 0 / 16
    const int n_sub = wave & 3;                          // 0..3
    const int arow  = m_sub + (lane & 15);
    const int aoff  = (lane < 16) ? 0 : 16;

    const v8h* bp = (const v8h*)(bfrag + (((size_t)n_sub * KCH) * 32 + lane) * 16);

    v8f acc = {};
#pragma unroll
    for (int kc = 0; kc < KCH; ++kc) {
        int kb = kc * 32;
        v8h a_lo = *(const v8h*)&sA[arow][kb + aoff];
        v8h a_hi = *(const v8h*)&sA[arow][kb + aoff + 8];
        v16h amat = __builtin_shufflevector(a_lo, a_hi,
            0,1,2,3,4,5,6,7,8,9,10,11,12,13,14,15);
        v8h b_lo = bp[0];
        v8h b_hi = bp[1];
        v16h bmat = __builtin_shufflevector(b_lo, b_hi,
            0,1,2,3,4,5,6,7,8,9,10,11,12,13,14,15);
        bp += 2 * 32;                                    // next K-chunk (32B * 32 lanes)
        __builtin_prefetch(bp, 0, 1);                    // global_prefetch next B frag
        acc = __builtin_amdgcn_wmma_f32_16x16x32_f16(
            false, amat, false, bmat, (short)0, acc, false, false);
    }

    const int n  = n_sub * 16 + (lane & 15);
    const float bv = bias[n];
    const int mbase = m_sub + ((lane >= 16) ? 8 : 0);
#pragma unroll
    for (int r = 0; r < 8; ++r) {
        int m = mbase + r;
        y[((size_t)(p0 + m) << 6) + n] = acc[r] + bv;
    }
}

// =============================================================
// 5) per-channel sum / sumsq reduction (y is [pixel][64])
// =============================================================
__global__ void k_reduce(const float* __restrict__ y, float* __restrict__ stats) {
    __shared__ float ls[256], ls2[256];
    int tid = threadIdx.x;
    int ch  = tid & 63;
    float s = 0.0f, s2 = 0.0f;
    size_t total  = (size_t)NPIX * OO;
    size_t stride = (size_t)gridDim.x * blockDim.x;      // multiple of 64
    for (size_t i = (size_t)blockIdx.x * blockDim.x + tid; i < total; i += stride) {
        float v = y[i];
        s += v; s2 += v * v;
    }
    ls[tid] = s; ls2[tid] = s2;
    __syncthreads();
    if (tid < 64) {
        float ts  = ls[tid]  + ls[tid + 64]  + ls[tid + 128]  + ls[tid + 192];
        float ts2 = ls2[tid] + ls2[tid + 64] + ls2[tid + 128] + ls2[tid + 192];
        atomicAdd(&stats[ch], ts);
        atomicAdd(&stats[64 + ch], ts2);
    }
}

// =============================================================
// 6) batch-norm + affine + SiLU, write NCHW output
// =============================================================
__global__ void k_final(const float* __restrict__ y, const float* __restrict__ stats,
                        const float* __restrict__ gamma, const float* __restrict__ beta,
                        float* __restrict__ out) {
    size_t idx = (size_t)blockIdx.x * blockDim.x + threadIdx.x;  // over NPIX*64
    int o = (int)(idx & 63);
    size_t p = idx >> 6;
    const float inv_n = 1.0f / (float)NPIX;
    float mean = stats[o] * inv_n;
    float var  = stats[64 + o] * inv_n - mean * mean;
    float rs   = rsqrtf(var + EPSV);
    float yn = (y[idx] - mean) * rs * gamma[o] + beta[o];
    float sl = yn / (1.0f + expf(-yn));
    int b  = (int)(p >> 14);
    int hw = (int)(p & (HWP - 1));
    out[(((size_t)b * OO + o) << 14) + hw] = sl;
}

// =============================================================
extern "C" void kernel_launch(void* const* d_in, const int* in_sizes, int n_in,
                              void* d_out, int out_size, void* d_ws, size_t ws_size,
                              hipStream_t stream) {
    const float* x      = (const float*)d_in[0];
    const float* weight = (const float*)d_in[1];
    const float* bias   = (const float*)d_in[2];
    const float* w_om   = (const float*)d_in[3];
    const float* b_om   = (const float*)d_in[4];
    const float* gamma  = (const float*)d_in[5];
    const float* beta   = (const float*)d_in[6];
    float* out = (float*)d_out;

    char* ws = (char*)d_ws;
    _Float16* xh    = (_Float16*)(ws + WS_XH);
    float*    om    = (float*)(ws + WS_OM);
    _Float16* bfrag = (_Float16*)(ws + WS_BF);
    float*    yb    = (float*)(ws + WS_Y);
    float*    stats = (float*)(ws + WS_STAT);

    k_init_stats<<<1, 128, 0, stream>>>(stats);
    k_nhwc<<<(BB * HWP * CC) / 256, 256, 0, stream>>>(x, xh);
    k_offconv<<<NPIX / 256, 256, 0, stream>>>(x, w_om, b_om, om);
    k_packB<<<(4 * KCH * 32 * 16 + 255) / 256, 256, 0, stream>>>(weight, bfrag);
    k_main<<<NPIX / 32, 256, 0, stream>>>(xh, om, bfrag, bias, yb);
    k_reduce<<<512, 256, 0, stream>>>(yb, stats);
    k_final<<<(NPIX * OO) / 256, 256, 0, stream>>>(yb, stats, gamma, beta, out);
}
